// GraphGatEdgeNet_55808805044432
// MI455X (gfx1250) — compile-verified, matrix operand order checked by reading, not run on hardware
//
#include <hip/hip_runtime.h>
#include <hip/hip_bf16.h>
#include <math.h>

// ---------------- constants ----------------
#define NN 20000
#define EE 100000
#define EPSBN 1e-5f

typedef __attribute__((ext_vector_type(16))) __bf16 v16bf;
typedef __attribute__((ext_vector_type(8)))  __bf16 v8bf;
typedef __attribute__((ext_vector_type(8)))  float  v8f;

// ---------------- helpers ----------------
__device__ inline __bf16 f2bf(float f) {
  unsigned u = __float_as_uint(f);
  unsigned r = u + 0x7FFFu + ((u >> 16) & 1u);   // round-to-nearest-even
  unsigned short h = (unsigned short)(r >> 16);
  __bf16 out;
  __builtin_memcpy(&out, &h, 2);
  return out;
}

__device__ inline v8f zero8() {
  v8f z;
#pragma unroll
  for (int i = 0; i < 8; ++i) z[i] = 0.f;
  return z;
}

// packed-pair K offset for 16-bit WMMA operand layout (ISA 7.12.2):
// VGPR j in 0..3 -> K = 2j + 8*half ; VGPR j in 4..7 -> K = 16 + 2(j-4) + 8*half
__device__ inline int kmap(int j, int half) {
  return (j < 4) ? (2 * j + 8 * half) : (16 + 2 * (j - 4) + 8 * half);
}

// A fragment: 16x32 bf16 tile, row-major in LDS (ld multiple of 8), k-offset k0.
// Lane layout collapses to two contiguous 8-element runs -> two ds_load_b128.
__device__ inline v16bf load_A_lds(const __bf16* A, int ld, int k0) {
  int lane = threadIdx.x & 31;
  int m = lane & 15, half = lane >> 4;
  const __bf16* base = A + m * ld + k0 + 8 * half;
  v8bf lo = *(const v8bf*)(base);
  v8bf hi = *(const v8bf*)(base + 16);
  return __builtin_shufflevector(lo, hi, 0, 1, 2, 3, 4, 5, 6, 7,
                                 8, 9, 10, 11, 12, 13, 14, 15);
}

// B fragment from fragment-major packed weights:
// fragment f = kt*ntiles + nt holds 32 lanes x 16 bf16 contiguous.
__device__ inline v16bf load_B_pk(const __bf16* Bp, int ntiles, int kt, int nt) {
  int lane = threadIdx.x & 31;
  return *(const v16bf*)(Bp + (((size_t)kt * ntiles + nt) * 32 + lane) * 16);
}

__device__ inline v8f wmma_bf16(v16bf a, v16bf b, v8f c) {
  return __builtin_amdgcn_wmma_f32_16x16x32_bf16(false, a, false, b, (short)0, c,
                                                 false, false);
}

__device__ inline float atomicMaxF(float* addr, float val) {
  unsigned* ua = (unsigned*)addr;
  unsigned old = __hip_atomic_load(ua, __ATOMIC_RELAXED, __HIP_MEMORY_SCOPE_AGENT);
  while (__uint_as_float(old) < val) {
    unsigned assumed = old;
    old = atomicCAS(ua, assumed, __float_as_uint(val));
    if (old == assumed) break;
  }
  return __uint_as_float(old);
}

// ---------------- small utility kernels ----------------
__global__ void k_fill(float* p, long n, float v) {
  long i = (long)blockIdx.x * blockDim.x + threadIdx.x;
  if (i < n) p[i] = v;
}

// Fold BN (optional) into fc weights and repack into WMMA fragment-major bf16.
// Thread per (fragment, lane). Also folds bias when bn present.
__global__ void k_pack_w(const float* __restrict__ w, int K, int ncols,
                         const float* g, const float* be, const float* m,
                         const float* v, const float* bias, __bf16* __restrict__ wp,
                         float* bout) {
  int idx = blockIdx.x * blockDim.x + threadIdx.x;
  int ntiles = ncols >> 4;
  int total = (K >> 5) * ntiles * 32;
  if (idx < total) {
    int lane = idx & 31;
    int f = idx >> 5;
    int nt = f % ntiles, kt = f / ntiles;
    int n = nt * 16 + (lane & 15);
    int half = lane >> 4;
    float s = g ? g[n] * rsqrtf(v[n] + EPSBN) : 1.f;
    __bf16* dst = wp + (size_t)idx * 16;
#pragma unroll
    for (int j = 0; j < 8; ++j) {
      int k = kt * 32 + kmap(j, half);
      dst[2 * j]     = f2bf(w[(size_t)k * ncols + n] * s);
      dst[2 * j + 1] = f2bf(w[(size_t)(k + 1) * ncols + n] * s);
    }
  }
  if (bout && idx < ncols) {
    float s = g[idx] * rsqrtf(v[idx] + EPSBN);
    bout[idx] = (bias[idx] - m[idx]) * s + be[idx];
  }
}

__global__ void k_mean_ea(const float* ea, float* meanEA) {
  int i = blockIdx.x * blockDim.x + threadIdx.x;
  if (i < EE * 64) atomicAdd(&meanEA[i & 63], ea[i] * (1.f / (float)EE));
}

// w_eff[c,h] = sum_cc gat_lin_edge_w[c, h*128+cc] * att_edge[h,cc]
__global__ void k_weff(const float* glew, const float* att_edge, float* weff) {
  int i = blockIdx.x * blockDim.x + threadIdx.x;
  if (i >= 64 * 8) return;
  int c = i / 8, h = i % 8;
  float s = 0.f;
  for (int cc = 0; cc < 128; ++cc)
    s += glew[c * 1024 + h * 128 + cc] * att_edge[h * 128 + cc];
  weff[c * 8 + h] = s;
}

__global__ void k_aeloop(const float* meanEA, const float* weff, float* aeLoop) {
  int h = threadIdx.x;
  if (h >= 8) return;
  float s = 0.f;
  for (int c = 0; c < 64; ++c) s += meanEA[c] * weff[c * 8 + h];
  aeLoop[h] = s;
}

// ---------------- WMMA node GEMM: C[M x ncols] = A[M x K] * Wpacked (+bias)
template <int K>
__global__ void k_gemm_nodes(const float* __restrict__ A, const __bf16* __restrict__ Wp,
                             const float* __restrict__ bias, float* __restrict__ C,
                             int M, int ncols) {
  __shared__ __bf16 As[16 * K];
  int tid = threadIdx.x;
  int m0 = blockIdx.x * 16;
  for (int idx = tid; idx < 16 * K; idx += 256) {
    int m = idx / K, c = idx % K;
    float v = (m0 + m < M) ? A[(size_t)(m0 + m) * K + c] : 0.f;
    As[idx] = f2bf(v);
  }
  __syncthreads();
  int wave = tid >> 5, lane = tid & 31;
  int ntiles = ncols >> 4;
  int ntile = blockIdx.y * 8 + wave;
  int n0 = ntile * 16;
  v8f acc = zero8();
#pragma unroll
  for (int kk = 0; kk < K; kk += 32)
    acc = wmma_bf16(load_A_lds(As, K, kk), load_B_pk(Wp, ntiles, kk >> 5, ntile), acc);
  int col = n0 + (lane & 15), half = lane >> 4;
  float bb = bias ? bias[col] : 0.f;
#pragma unroll
  for (int r = 0; r < 8; ++r) {
    int m = m0 + r + 8 * half;
    if (m < M) C[(size_t)m * ncols + col] = acc[r] + bb;
  }
}

// ---------------- attention scalars ----------------
__global__ void k_att_dots(const float* xw, const float* aw_s, const float* aw_d,
                           float* a_src, float* a_dst) {
  int i = blockIdx.x * blockDim.x + threadIdx.x;
  if (i >= NN * 8) return;
  int n = i / 8, h = i % 8;
  const float* row = xw + (size_t)n * 1024 + h * 128;
  float s = 0.f, d = 0.f;
  for (int c = 0; c < 128; ++c) {
    s += row[c] * aw_s[h * 128 + c];
    d += row[c] * aw_d[h * 128 + c];
  }
  a_src[i] = s;
  a_dst[i] = d;
}

__global__ void k_edge_logits(const float* ea, const float* weff, const float* a_src,
                              const float* a_dst, const int* src, const int* dst,
                              float* logit, float* mx) {
  int i = blockIdx.x * blockDim.x + threadIdx.x;
  if (i >= EE * 8) return;
  int e = i / 8, h = i % 8;
  float aev = 0.f;
  for (int c = 0; c < 64; ++c) aev += ea[e * 64 + c] * weff[c * 8 + h];
  int s = src[e], d = dst[e];
  float l = a_src[s * 8 + h] + a_dst[d * 8 + h] + aev;
  l = (l > 0.f) ? l : 0.2f * l;
  logit[i] = l;
  atomicMaxF(&mx[d * 8 + h], l);
}

__global__ void k_loop_logits(const float* a_src, const float* a_dst,
                              const float* aeLoop, float* llog, float* mx) {
  int i = blockIdx.x * blockDim.x + threadIdx.x;
  if (i >= NN * 8) return;
  int h = i % 8;
  float l = a_src[i] + a_dst[i] + aeLoop[h];
  l = (l > 0.f) ? l : 0.2f * l;
  llog[i] = l;
  atomicMaxF(&mx[i], l);
}

__global__ void k_edge_alpha(float* logit, const float* mx, float* sm, const int* dst) {
  int i = blockIdx.x * blockDim.x + threadIdx.x;
  if (i >= EE * 8) return;
  int e = i / 8, h = i % 8;
  int d = dst[e];
  float ex = expf(logit[i] - mx[d * 8 + h]);
  logit[i] = ex;
  atomicAdd(&sm[d * 8 + h], ex);
}

__global__ void k_loop_alpha(float* llog, const float* mx, float* sm) {
  int i = blockIdx.x * blockDim.x + threadIdx.x;
  if (i >= NN * 8) return;
  float ex = expf(llog[i] - mx[i]);
  llog[i] = ex;
  atomicAdd(&sm[i], ex);
}

__global__ void k_edge_agg(const float* xw, const float* alphaE, const float* sm,
                           const int* src, const int* dst, float* agg) {
  int tid = threadIdx.x;
  int e = blockIdx.x * 2 + (tid >> 7);
  int c = tid & 127;
  if (e >= EE) return;
  int s = src[e], d = dst[e];
  for (int h = 0; h < 8; ++h) {
    float coef = alphaE[e * 8 + h] / (sm[d * 8 + h] + 1e-16f);
    atomicAdd(&agg[(size_t)d * 1024 + h * 128 + c],
              xw[(size_t)s * 1024 + h * 128 + c] * coef);
  }
}

__global__ void k_loop_agg(const float* xw, const float* llog, const float* sm,
                           float* agg) {
  int i = blockIdx.x * blockDim.x + threadIdx.x;
  if (i >= NN * 128) return;
  int n = i >> 7, c = i & 127;
  for (int h = 0; h < 8; ++h) {
    float coef = llog[n * 8 + h] / (sm[n * 8 + h] + 1e-16f);
    agg[(size_t)n * 1024 + h * 128 + c] += xw[(size_t)n * 1024 + h * 128 + c] * coef;
  }
}

__global__ void k_bn_relu_g1(float* agg, const float* bias, const float* g,
                             const float* b, const float* m, const float* v) {
  long i = (long)blockIdx.x * blockDim.x + threadIdx.x;
  if (i >= (long)NN * 1024) return;
  int j = (int)(i & 1023);
  float val = agg[i] + bias[j];
  float s = g[j] * rsqrtf(v[j] + EPSBN);
  val = (val - m[j]) * s + b[j];
  agg[i] = fmaxf(val, 0.f);
}

__global__ void k_count(const int* dst, float* cnt) {
  int i = blockIdx.x * blockDim.x + threadIdx.x;
  if (i < EE) atomicAdd(&cnt[dst[i]], 1.f);
}

// ---------------- WMMA EdgeConv: per-edge 2-layer MLP with folded BN, scatter-add
// z_e = [x[dst], x[src]-x[dst]] (2F) ; h1 = relu(z@w1f + b1f) ; h2 = relu(h1@w2f + b2f)
// accum[dst] += h2
template <int F>
__global__ void k_edgeconv(const float* __restrict__ xin, const int* __restrict__ src,
                           const int* __restrict__ dst, const __bf16* __restrict__ w1,
                           const float* __restrict__ b1, const __bf16* __restrict__ w2,
                           const float* __restrict__ b2, float* __restrict__ accum,
                           int nE) {
  constexpr int K1 = 2 * F;
  constexpr int KCH = (K1 < 256) ? K1 : 256;
  __shared__ __bf16 As[16 * KCH];
  __shared__ __bf16 Hs[16 * 128];
  __shared__ int sS[16], sD[16];
  int tid = threadIdx.x, lane = tid & 31, wave = tid >> 5;
  int e0 = blockIdx.x * 16;
  if (tid < 16) {
    int e = e0 + tid;
    sS[tid] = (e < nE) ? src[e] : 0;
    sD[tid] = (e < nE) ? dst[e] : -1;
  }
  __syncthreads();
  v8f acc = zero8();
  for (int kc = 0; kc < K1; kc += KCH) {
    __syncthreads();
    for (int idx = tid; idx < 16 * KCH; idx += 256) {
      int m = idx / KCH, cl = idx % KCH;
      int c = kc + cl;
      int d = (sD[m] < 0) ? 0 : sD[m];
      int s = sS[m];
      float v;
      if (c < F) v = xin[(size_t)d * F + c];
      else {
        int cc = c - F;
        v = xin[(size_t)s * F + cc] - xin[(size_t)d * F + cc];
      }
      As[m * KCH + cl] = f2bf(v);
    }
    __syncthreads();
#pragma unroll
    for (int kk = 0; kk < KCH; kk += 32)
      acc = wmma_bf16(load_A_lds(As, KCH, kk),
                      load_B_pk(w1, 8, (kc + kk) >> 5, wave), acc);
  }
  int col = wave * 16 + (lane & 15), half = lane >> 4;
  float bb = b1[col];
#pragma unroll
  for (int r = 0; r < 8; ++r) {
    int m = r + 8 * half;
    Hs[m * 128 + col] = f2bf(fmaxf(acc[r] + bb, 0.f));
  }
  __syncthreads();
  v8f acc2 = zero8();
#pragma unroll
  for (int kk = 0; kk < 128; kk += 32)
    acc2 = wmma_bf16(load_A_lds(Hs, 128, kk), load_B_pk(w2, 8, kk >> 5, wave), acc2);
  float bb2 = b2[col];
#pragma unroll
  for (int r = 0; r < 8; ++r) {
    int m = r + 8 * half;
    int d = sD[m];
    float v = fmaxf(acc2[r] + bb2, 0.f);
    if (d >= 0) atomicAdd(&accum[(size_t)d * 128 + col], v);
  }
}

// mean-aggregate + optional residual + optional BN + optional relu
__global__ void k_finalize(const float* accum, const float* cnt, const float* g,
                           const float* b, const float* m, const float* v,
                           const float* res, float* out, int relu) {
  int i = blockIdx.x * blockDim.x + threadIdx.x;
  if (i >= NN * 128) return;
  int n = i >> 7, j = i & 127;
  float val = accum[i] / fmaxf(cnt[n], 1.f);
  if (res) val += res[i];
  if (g) {
    float s = g[j] * rsqrtf(v[j] + EPSBN);
    val = (val - m[j]) * s + b[j];
  }
  if (relu) val = fmaxf(val, 0.f);
  out[i] = val;
}

__global__ void k_fc(const float* g4, const float* fw, const float* fb, float* out) {
  int i = blockIdx.x * blockDim.x + threadIdx.x;
  if (i >= NN * 2) return;
  int n = i >> 1, o = i & 1;
  float s = fb[o];
  for (int c = 0; c < 128; ++c) s += g4[n * 128 + c] * fw[c * 2 + o];
  out[i] = s;
}

// ---------------- host ----------------
static inline unsigned gsz(long n) { return (unsigned)((n + 255) / 256); }

extern "C" void kernel_launch(void* const* d_in, const int* in_sizes, int n_in,
                              void* d_out, int out_size, void* d_ws, size_t ws_size,
                              hipStream_t stream) {
  // inputs in setup_inputs() dict order (recursive insertion order)
  const float* x        = (const float*)d_in[0];
  const int*   ei       = (const int*)d_in[1];
  const float* ea       = (const float*)d_in[2];
  const float* av_w     = (const float*)d_in[3];
  const float* av_b     = (const float*)d_in[4];
  const float* glw      = (const float*)d_in[5];
  const float* glew     = (const float*)d_in[6];
  const float* att_src  = (const float*)d_in[7];
  const float* att_dst  = (const float*)d_in[8];
  const float* att_edge = (const float*)d_in[9];
  const float* gat_bias = (const float*)d_in[10];
  const float* bn1g = (const float*)d_in[11], *bn1b = (const float*)d_in[12];
  const float* bn1m = (const float*)d_in[13], *bn1v = (const float*)d_in[14];
  const float* m2w1 = (const float*)d_in[15], *m2b1 = (const float*)d_in[16];
  const float* m2n1g = (const float*)d_in[17], *m2n1b = (const float*)d_in[18];
  const float* m2n1m = (const float*)d_in[19], *m2n1v = (const float*)d_in[20];
  const float* m2w2 = (const float*)d_in[21], *m2b2 = (const float*)d_in[22];
  const float* m2n2g = (const float*)d_in[23], *m2n2b = (const float*)d_in[24];
  const float* m2n2m = (const float*)d_in[25], *m2n2v = (const float*)d_in[26];
  const float* bn2g = (const float*)d_in[27], *bn2b = (const float*)d_in[28];
  const float* bn2m = (const float*)d_in[29], *bn2v = (const float*)d_in[30];
  const float* m3w1 = (const float*)d_in[31], *m3b1 = (const float*)d_in[32];
  const float* m3n1g = (const float*)d_in[33], *m3n1b = (const float*)d_in[34];
  const float* m3n1m = (const float*)d_in[35], *m3n1v = (const float*)d_in[36];
  const float* m3w2 = (const float*)d_in[37], *m3b2 = (const float*)d_in[38];
  const float* m3n2g = (const float*)d_in[39], *m3n2b = (const float*)d_in[40];
  const float* m3n2m = (const float*)d_in[41], *m3n2v = (const float*)d_in[42];
  const float* bn3g = (const float*)d_in[43], *bn3b = (const float*)d_in[44];
  const float* bn3m = (const float*)d_in[45], *bn3v = (const float*)d_in[46];
  const float* m4w1 = (const float*)d_in[47], *m4b1 = (const float*)d_in[48];
  const float* m4n1g = (const float*)d_in[49], *m4n1b = (const float*)d_in[50];
  const float* m4n1m = (const float*)d_in[51], *m4n1v = (const float*)d_in[52];
  const float* m4w2 = (const float*)d_in[53], *m4b2 = (const float*)d_in[54];
  const float* m4n2g = (const float*)d_in[55], *m4n2b = (const float*)d_in[56];
  const float* m4n2m = (const float*)d_in[57], *m4n2v = (const float*)d_in[58];
  const float* fc_w = (const float*)d_in[59], *fc_b = (const float*)d_in[60];

  const int* srcI = ei;
  const int* dstI = ei + EE;

  // workspace layout (f32 region, then 32B-aligned bf16 packed-weight region)
  float* W = (float*)d_ws;
  size_t o = 0;
  auto A = [&](size_t n) { float* p = W + o; o += n; return p; };
  float* gf     = A((size_t)NN * 128);
  float* xw     = A((size_t)NN * 1024);
  float* agg    = A((size_t)NN * 1024);   // becomes g1 in place
  float* a_src  = A((size_t)NN * 8);
  float* a_dst  = A((size_t)NN * 8);
  float* mx     = A((size_t)NN * 8);
  float* sm     = A((size_t)NN * 8);
  float* logit  = A((size_t)EE * 8);      // then holds exp(alpha) in place
  float* llog   = A((size_t)NN * 8);
  float* meanEA = A(64);
  float* aeLoop = A(8);
  float* weff   = A(64 * 8);
  float* cnt    = A(NN);
  float* accum  = A((size_t)NN * 128);
  float* g2     = A((size_t)NN * 128);
  float* g3     = A((size_t)NN * 128);
  float* g4     = A((size_t)NN * 128);
  float* b1f2 = A(128); float* b2f2 = A(128);
  float* b1f3 = A(128); float* b2f3 = A(128);
  float* b1f4 = A(128); float* b2f4 = A(128);
  __bf16* Bbase = (__bf16*)(W + o);
  size_t ob = 0;
  auto AB = [&](size_t n) { __bf16* p = Bbase + ob; ob += n; return p; };
  __bf16* avwB = AB(256 * 128);
  __bf16* glwB = AB(128 * 1024);
  __bf16* w1f2 = AB(2048 * 128);
  __bf16* w2f2 = AB(128 * 128);
  __bf16* w1f3 = AB(256 * 128);
  __bf16* w2f3 = AB(128 * 128);
  __bf16* w1f4 = AB(256 * 128);
  __bf16* w2f4 = AB(128 * 128);

  // ---- init accumulators ----
  k_fill<<<gsz(64), 256, 0, stream>>>(meanEA, 64, 0.f);
  k_fill<<<gsz((long)NN * 8), 256, 0, stream>>>(mx, (long)NN * 8, -1e30f);
  k_fill<<<gsz((long)NN * 8), 256, 0, stream>>>(sm, (long)NN * 8, 0.f);
  k_fill<<<gsz((long)NN * 1024), 256, 0, stream>>>(agg, (long)NN * 1024, 0.f);
  k_fill<<<gsz(NN), 256, 0, stream>>>(cnt, NN, 0.f);

  // ---- weight prep: fold BN + repack to fragment-major bf16 ----
  k_mean_ea<<<gsz((long)EE * 64), 256, 0, stream>>>(ea, meanEA);
  k_weff<<<2, 256, 0, stream>>>(glew, att_edge, weff);
  k_aeloop<<<1, 8, 0, stream>>>(meanEA, weff, aeLoop);
  k_pack_w<<<gsz(8 * 8 * 32), 256, 0, stream>>>(av_w, 256, 128, nullptr, nullptr,
                                                nullptr, nullptr, nullptr, avwB, nullptr);
  k_pack_w<<<gsz(4 * 64 * 32), 256, 0, stream>>>(glw, 128, 1024, nullptr, nullptr,
                                                 nullptr, nullptr, nullptr, glwB, nullptr);
  k_pack_w<<<gsz(64 * 8 * 32), 256, 0, stream>>>(m2w1, 2048, 128, m2n1g, m2n1b, m2n1m,
                                                 m2n1v, m2b1, w1f2, b1f2);
  k_pack_w<<<gsz(4 * 8 * 32), 256, 0, stream>>>(m2w2, 128, 128, m2n2g, m2n2b, m2n2m,
                                                m2n2v, m2b2, w2f2, b2f2);
  k_pack_w<<<gsz(8 * 8 * 32), 256, 0, stream>>>(m3w1, 256, 128, m3n1g, m3n1b, m3n1m,
                                                m3n1v, m3b1, w1f3, b1f3);
  k_pack_w<<<gsz(4 * 8 * 32), 256, 0, stream>>>(m3w2, 128, 128, m3n2g, m3n2b, m3n2m,
                                                m3n2v, m3b2, w2f3, b2f3);
  k_pack_w<<<gsz(8 * 8 * 32), 256, 0, stream>>>(m4w1, 256, 128, m4n1g, m4n1b, m4n1m,
                                                m4n1v, m4b1, w1f4, b1f4);
  k_pack_w<<<gsz(4 * 8 * 32), 256, 0, stream>>>(m4w2, 128, 128, m4n2g, m4n2b, m4n2m,
                                                m4n2v, m4b2, w2f4, b2f4);

  // ---- gf = x.reshape(N,256) @ av_w + av_b  (WMMA) ----
  k_gemm_nodes<256><<<dim3(1250, 1), 256, 0, stream>>>(x, avwB, av_b, gf, NN, 128);
  // ---- xw = gf @ gat_lin_w  (WMMA) ----
  k_gemm_nodes<128><<<dim3(1250, 8), 256, 0, stream>>>(gf, glwB, nullptr, xw, NN, 1024);

  // ---- GAT attention (segment softmax over dst, incl. self-loops) ----
  k_att_dots<<<gsz((long)NN * 8), 256, 0, stream>>>(xw, att_src, att_dst, a_src, a_dst);
  k_edge_logits<<<gsz((long)EE * 8), 256, 0, stream>>>(ea, weff, a_src, a_dst, srcI,
                                                       dstI, logit, mx);
  k_loop_logits<<<gsz((long)NN * 8), 256, 0, stream>>>(a_src, a_dst, aeLoop, llog, mx);
  k_edge_alpha<<<gsz((long)EE * 8), 256, 0, stream>>>(logit, mx, sm, dstI);
  k_loop_alpha<<<gsz((long)NN * 8), 256, 0, stream>>>(llog, mx, sm);
  k_edge_agg<<<EE / 2, 256, 0, stream>>>(xw, logit, sm, srcI, dstI, agg);
  k_loop_agg<<<gsz((long)NN * 128), 256, 0, stream>>>(xw, llog, sm, agg);
  k_bn_relu_g1<<<gsz((long)NN * 1024), 256, 0, stream>>>(agg, gat_bias, bn1g, bn1b,
                                                         bn1m, bn1v);

  // ---- EdgeConv layers (WMMA, BN folded) ----
  k_count<<<gsz(EE), 256, 0, stream>>>(dstI, cnt);

  k_fill<<<gsz((long)NN * 128), 256, 0, stream>>>(accum, (long)NN * 128, 0.f);
  k_edgeconv<1024><<<EE / 16, 256, 0, stream>>>(agg, srcI, dstI, w1f2, b1f2, w2f2,
                                                b2f2, accum, EE);
  k_finalize<<<gsz((long)NN * 128), 256, 0, stream>>>(accum, cnt, bn2g, bn2b, bn2m,
                                                      bn2v, nullptr, g2, 1);

  k_fill<<<gsz((long)NN * 128), 256, 0, stream>>>(accum, (long)NN * 128, 0.f);
  k_edgeconv<128><<<EE / 16, 256, 0, stream>>>(g2, srcI, dstI, w1f3, b1f3, w2f3, b2f3,
                                               accum, EE);
  k_finalize<<<gsz((long)NN * 128), 256, 0, stream>>>(accum, cnt, bn3g, bn3b, bn3m,
                                                      bn3v, g2, g3, 1);

  k_fill<<<gsz((long)NN * 128), 256, 0, stream>>>(accum, (long)NN * 128, 0.f);
  k_edgeconv<128><<<EE / 16, 256, 0, stream>>>(g3, srcI, dstI, w1f4, b1f4, w2f4, b2f4,
                                               accum, EE);
  k_finalize<<<gsz((long)NN * 128), 256, 0, stream>>>(accum, cnt, nullptr, nullptr,
                                                      nullptr, nullptr, g3, g4, 0);

  // ---- final fc ----
  k_fc<<<gsz((long)NN * 2), 256, 0, stream>>>(g4, fc_w, fc_b, (float*)d_out);
}
